// AsymmetricLossCustomMS_18064632447149
// MI455X (gfx1250) — compile-verified, matrix operand order checked by reading, not run on hardware
//
#include <hip/hip_runtime.h>

#define B_N   2048
#define C_N   9605
#define L_N   8
#define TOPK_N 16

typedef __attribute__((ext_vector_type(16))) _Float16 v16h;
typedef __attribute__((ext_vector_type(8)))  float    v8f;
typedef __attribute__((ext_vector_type(4)))  unsigned u32x4;
typedef __attribute__((ext_vector_type(8)))  int      i32x8;
typedef __attribute__((ext_vector_type(4)))  int      i32x4;

__device__ __forceinline__ float sigmoidf_(float v) {
    return 1.0f / (1.0f + __expf(-v));
}

// our_rank_loss: d = x2-x1+margin; s = sigmoid(5d); 2s if violated (d>0) else s
__device__ __forceinline__ float rank_loss_(float x1, float x2) {
    float d = x2 - x1 + 0.05f;
    float s = 1.0f / (1.0f + __expf(-5.0f * d));
    return d > 0.0f ? 2.0f * s : s;
}

// ---- wave32-aware block reductions (256 threads = 8 waves) ----
__device__ __forceinline__ float blockMaxF_(float v, float* s, unsigned tid) {
    #pragma unroll
    for (int o = 16; o; o >>= 1) v = fmaxf(v, __shfl_xor(v, o));
    if ((tid & 31u) == 0u) s[tid >> 5] = v;
    __syncthreads();
    if (tid == 0u) {
        float m = s[0];
        #pragma unroll
        for (int i = 1; i < 8; ++i) m = fmaxf(m, s[i]);
        s[0] = m;
    }
    __syncthreads();
    float r = s[0];
    __syncthreads();
    return r;
}

__device__ __forceinline__ unsigned blockMaxU_(unsigned v, unsigned* s, unsigned tid) {
    #pragma unroll
    for (int o = 16; o; o >>= 1) { unsigned t = __shfl_xor(v, o); v = t > v ? t : v; }
    if ((tid & 31u) == 0u) s[tid >> 5] = v;
    __syncthreads();
    if (tid == 0u) {
        unsigned m = s[0];
        #pragma unroll
        for (int i = 1; i < 8; ++i) m = s[i] > m ? s[i] : m;
        s[0] = m;
    }
    __syncthreads();
    unsigned r = s[0];
    __syncthreads();
    return r;
}

__device__ __forceinline__ int blockAddI_(int v, int* s, unsigned tid) {
    #pragma unroll
    for (int o = 16; o; o >>= 1) v += __shfl_xor(v, o);
    if ((tid & 31u) == 0u) s[tid >> 5] = v;
    __syncthreads();
    if (tid == 0u) {
        int m = 0;
        #pragma unroll
        for (int i = 0; i < 8; ++i) m += s[i];
        s[0] = m;
    }
    __syncthreads();
    int r = s[0];
    __syncthreads();
    return r;
}

// ---- kernel 0: pack [L,C] bool mask into per-class 8-bit group bitmask ----
__global__ void __launch_bounds__(256) k_pack_groups(const unsigned char* __restrict__ gm,
                                                     unsigned char* __restrict__ gbits) {
    int c = blockIdx.x * 256 + threadIdx.x;
    if (c >= C_N) return;
    unsigned bits = 0;
    #pragma unroll
    for (int l = 0; l < L_N; ++l) bits |= (gm[(size_t)l * C_N + c] ? 1u : 0u) << l;
    gbits[c] = (unsigned char)bits;
}

// ---- kernel 1: gt = (Y @ Mᵀ) > 0 and gt_neg via v_wmma_f32_16x16x32_f16 ----
// One wave per 16-row tile. A: 16x32 f16 per ISA layout (lane m=lane&15,
// kb=(lane>>4)*8; v0..v3 hold K=kb..kb+7, v4..v7 hold K=16+kb..16+kb+7).
// B: 32x16 mirror layout with N=lane&15, built from packed gbits (0/1).
// D layout: VGPR r holds rows r (lanes0-15) and 8+r (lanes16-31) -> ballot.
__global__ void __launch_bounds__(256) k_group_hits_wmma(const float* __restrict__ y,
                                                         const float* __restrict__ yn,
                                                         const unsigned char* __restrict__ gbits,
                                                         unsigned char* __restrict__ gt8,
                                                         unsigned char* __restrict__ gtneg8) {
    const unsigned lane = threadIdx.x & 31u;
    const unsigned wave = threadIdx.x >> 5;
    const unsigned tile = blockIdx.x * 8u + wave;
    const unsigned row0 = tile * 16u;
    if (row0 >= (unsigned)B_N) return;

    const unsigned m  = lane & 15u;        // A-matrix row / B-matrix col
    const unsigned kb = (lane >> 4) * 8u;  // K sub-base per half-wave
    const float* yrow  = y  + (size_t)(row0 + m) * C_N;
    const float* ynrow = yn + (size_t)(row0 + m) * C_N;

    v8f accY = {};
    v8f accN = {};
    const int chunks = (C_N + 31) / 32;
    for (int ch = 0; ch < chunks; ++ch) {
        const int k0 = ch * 32;
        v16h a, an, bm;
        #pragma unroll
        for (int j = 0; j < 16; ++j) {
            int K = (j < 8) ? (int)(kb + (unsigned)j) : (int)(16u + kb + (unsigned)(j - 8));
            int k = k0 + K;
            float av = 0.0f, anv = 0.0f;
            unsigned gb = 0;
            if (k < C_N) { av = yrow[k]; anv = ynrow[k]; gb = gbits[k]; }
            a[j]  = (_Float16)av;
            an[j] = (_Float16)anv;
            bm[j] = (_Float16)(float)((m < 8u) ? ((gb >> m) & 1u) : 0u);
        }
        accY = __builtin_amdgcn_wmma_f32_16x16x32_f16(false, a,  false, bm, (short)0, accY, false, false);
        accN = __builtin_amdgcn_wmma_f32_16x16x32_f16(false, an, false, bm, (short)0, accN, false, false);
    }

    unsigned myGt = 0, myGn = 0;
    #pragma unroll
    for (int r = 0; r < 8; ++r) {
        unsigned balY = (unsigned)__ballot(accY[r] > 0.5f);
        unsigned balN = (unsigned)__ballot(accN[r] > 0.5f);
        if (lane == (unsigned)r)       { myGt = balY & 0xFFu;          myGn = balN & 0xFFu; }
        if (lane == (unsigned)(8 + r)) { myGt = (balY >> 16) & 0xFFu;  myGn = (balN >> 16) & 0xFFu; }
    }
    if (lane < 16u) {
        gt8[row0 + lane]    = (unsigned char)myGt;
        gtneg8[row0 + lane] = (unsigned char)myGn;
    }
}

// ---- kernel 2: per-row streaming pass + exact top-16 threshold + loss ----
__global__ void __launch_bounds__(256) k_row_loss(const float* __restrict__ x,
                                                  const unsigned char* __restrict__ gbits,
                                                  const unsigned char* __restrict__ gt8,
                                                  const unsigned char* __restrict__ gtneg8,
                                                  float* __restrict__ losses) {
    __shared__ unsigned lds_k[C_N];   // raw f32 bits, then monotone keys (38.4 KB)
    __shared__ float    sf[8];
    __shared__ unsigned su[8];
    __shared__ int      si[8];

    const unsigned tid = threadIdx.x;
    const unsigned b   = blockIdx.x;
    const float* xrow  = x + (size_t)b * C_N;

#if __has_builtin(__builtin_amdgcn_tensor_load_to_lds) && __has_builtin(__builtin_amdgcn_s_wait_tensorcnt)
    // TDM bulk DMA of the row into LDS (wave 0 issues; EXEC ignored by TDM).
    if ((tid >> 5) == 0u) {
        unsigned lds_off = (unsigned)(unsigned long long)(const void*)lds_k; // low32 of generic = LDS offset
        unsigned long long ga = (unsigned long long)(const void*)xrow;
        u32x4 g0;
        g0[0] = 1u;                                                   // count=1 (valid), user mode
        g0[1] = lds_off;                                              // lds_addr
        g0[2] = (unsigned)ga;                                         // global_addr[31:0]
        g0[3] = ((unsigned)((ga >> 32) & 0x01FFFFFFull)) | (2u << 30);// global_addr[56:32] | type=2
        i32x8 g1;
        g1[0] = (int)0x00020000u;                            // wg_mask=0, data_size=2 (4B)
        g1[1] = (int)(((unsigned)C_N & 0xFFFFu) << 16);      // tensor_dim0[15:0]
        g1[2] = (int)((((unsigned)C_N >> 16) & 0xFFFFu) | (1u << 16)); // tensor_dim0[31:16] | tensor_dim1=1
        g1[3] = (int)((unsigned)C_N << 16);                  // tile_dim0 = C
        g1[4] = (int)1;                                      // tile_dim1=1, tile_dim2=0
        g1[5] = (int)C_N;                                    // tensor_dim0_stride lo
        g1[6] = 0;
        g1[7] = 0;
        i32x4 gz4; gz4[0] = 0; gz4[1] = 0; gz4[2] = 0; gz4[3] = 0;
        i32x8 gz8;
        #pragma unroll
        for (int i = 0; i < 8; ++i) gz8[i] = 0;
        __builtin_amdgcn_tensor_load_to_lds(g0, g1, gz4, gz4, gz8, 0);
        __builtin_amdgcn_s_wait_tensorcnt(0);
    }
#else
    for (unsigned c = tid; c < (unsigned)C_N; c += 256u)
        lds_k[c] = __float_as_uint(xrow[c]);
#endif
    __syncthreads();

    // Streaming pass: per-group max of raw x, union max, transform bits->keys.
    float gmax[L_N];
    #pragma unroll
    for (int l = 0; l < L_N; ++l) gmax[l] = -__builtin_inff();
    float umax = -__builtin_inff();
    for (unsigned c = tid; c < (unsigned)C_N; c += 256u) {
        unsigned u = lds_k[c];
        float xv = __uint_as_float(u);
        unsigned g = gbits[c];
        if (g) {
            umax = fmaxf(umax, xv);
            #pragma unroll
            for (int l = 0; l < L_N; ++l)
                if ((g >> l) & 1u) gmax[l] = fmaxf(gmax[l], xv);
        }
        lds_k[c] = (u & 0x80000000u) ? ~u : (u | 0x80000000u); // order-preserving (ascending) monotone key
    }
    __syncthreads();

    #pragma unroll
    for (int l = 0; l < L_N; ++l) gmax[l] = blockMaxF_(gmax[l], sf, tid);
    umax = blockMaxF_(umax, sf, tid);

    // Exact 16th-largest via iterative (max-below-bound, tie count).
    unsigned bound = 0xFFFFFFFFu;
    unsigned kth = 0u;
    int remaining = TOPK_N;
    for (int round = 0; round < TOPK_N && remaining > 0; ++round) {
        unsigned lmax = 0u;
        for (unsigned c = tid; c < (unsigned)C_N; c += 256u) {
            unsigned k = lds_k[c];
            if (k < bound && k > lmax) lmax = k;
        }
        unsigned mkey = blockMaxU_(lmax, su, tid);
        int lcnt = 0;
        for (unsigned c = tid; c < (unsigned)C_N; c += 256u)
            lcnt += (lds_k[c] == mkey) ? 1 : 0;
        int cnt = blockAddI_(lcnt, si, tid);
        if (cnt == 0) break;     // degenerate safety
        remaining -= cnt;
        bound = mkey;
        kth = mkey;
    }

    if (tid == 0u) {
        unsigned uk  = (kth & 0x80000000u) ? (kth & 0x7FFFFFFFu) : ~kth;
        float x16    = __uint_as_float(uk);
        float thres  = fmaxf(sigmoidf_(x16), 0.3f);   // max(16th sigmoid, alpha_other)

        unsigned gtb = gt8[b];
        unsigned gnb = gtneg8[b];
        float caseB = 0.0f;
        float negx  = -__builtin_inff();
        #pragma unroll
        for (int l = 0; l < L_N; ++l) {
            float gs = sigmoidf_(gmax[l]);            // sigmoid(max x) == max sigmoid
            caseB += ((gtb >> l) & 1u) ? rank_loss_(gs, thres)
                                       : rank_loss_(thres, gs);
            if ((gnb >> l) & 1u) negx = fmaxf(negx, gmax[l]);
        }
        float us       = sigmoidf_(umax);
        float negscore = gnb ? sigmoidf_(negx) : 0.0f;
        float caseA    = 0.5f * rank_loss_(thres, us) + 0.5f * rank_loss_(thres, negscore);
        losses[b] = gtb ? caseB : caseA;
    }
}

// ---- kernel 3: deterministic mean over B rows ----
__global__ void __launch_bounds__(256) k_mean(const float* __restrict__ losses,
                                              float* __restrict__ out) {
    __shared__ float sp[8];
    unsigned tid = threadIdx.x;
    float s = 0.0f;
    #pragma unroll
    for (int i = 0; i < B_N / 256; ++i) s += losses[tid + (unsigned)i * 256u];
    #pragma unroll
    for (int o = 16; o; o >>= 1) s += __shfl_xor(s, o);
    if ((tid & 31u) == 0u) sp[tid >> 5] = s;
    __syncthreads();
    if (tid == 0u) {
        float t = 0.0f;
        #pragma unroll
        for (int i = 0; i < 8; ++i) t += sp[i];
        out[0] = t / (float)B_N;
    }
}

extern "C" void kernel_launch(void* const* d_in, const int* in_sizes, int n_in,
                              void* d_out, int out_size, void* d_ws, size_t ws_size,
                              hipStream_t stream) {
    (void)in_sizes; (void)n_in; (void)out_size; (void)ws_size;
    const float*         x   = (const float*)d_in[0];
    const float*         y   = (const float*)d_in[1];
    const float*         yn  = (const float*)d_in[2];
    const unsigned char* gm  = (const unsigned char*)d_in[3]; // jax bool -> 1 byte/elem
    float*               out = (float*)d_out;

    // workspace layout
    float*         losses  = (float*)d_ws;                                  // B floats
    unsigned char* gbits   = (unsigned char*)d_ws + B_N * sizeof(float);    // C bytes
    unsigned char* gt8     = gbits + ((C_N + 127) / 128) * 128;             // B bytes
    unsigned char* gtneg8  = gt8 + B_N;                                     // B bytes

    k_pack_groups<<<(C_N + 255) / 256, 256, 0, stream>>>(gm, gbits);
    k_group_hits_wmma<<<(B_N / 16) / 8, 256, 0, stream>>>(y, yn, gbits, gt8, gtneg8);
    k_row_loss<<<B_N, 256, 0, stream>>>(x, gbits, gt8, gtneg8, losses);
    k_mean<<<1, 256, 0, stream>>>(losses, out);
}
